// DUAN_25168508355367
// MI455X (gfx1250) — matrix operand order repaired and checked by reading, not compiled, not run on hardware
//
#include <hip/hip_runtime.h>
#include <hip/hip_bf16.h>
#include <math.h>

typedef __attribute__((ext_vector_type(16))) _Float16 v16h;
typedef __attribute__((ext_vector_type(8)))  _Float16 v8h;
typedef __attribute__((ext_vector_type(2)))  _Float16 h2;
typedef __attribute__((ext_vector_type(8)))  float    v8f;

#define TB 8
#define TC 512
#define TL 8192
#define TH 128
#define LT 128
#define KKEEP 358   // max(1, int(512*0.7))

// ---------------------------------------------------------------------------
// K1: per-(b,c) sums of x, x^2, c over L; also zero the g_mix accumulator.
// ---------------------------------------------------------------------------
__launch_bounds__(256)
__global__ void duan_stats_kernel(const float* __restrict__ x,
                                  const float* __restrict__ cmat,
                                  float* __restrict__ sumx,
                                  float* __restrict__ sumx2,
                                  float* __restrict__ sumc,
                                  float* __restrict__ gmix_zero) {
    const int bc = blockIdx.y * TC + blockIdx.x;
    const size_t base = (size_t)bc * TL;
    const float4* x4 = (const float4*)(x + base);
    const float4* c4 = (const float4*)(cmat + base);
    float sx = 0.f, sxx = 0.f, sc = 0.f;
    for (int i = threadIdx.x; i < TL / 4; i += 256) {
        __builtin_prefetch(&x4[i + 512], 0, 1);
        __builtin_prefetch(&c4[i + 512], 0, 1);
        float4 v = x4[i];
        sx  += v.x + v.y + v.z + v.w;
        sxx += v.x * v.x + v.y * v.y + v.z * v.z + v.w * v.w;
        float4 u = c4[i];
        sc  += u.x + u.y + u.z + u.w;
    }
    __shared__ float red[3][256];
    red[0][threadIdx.x] = sx;
    red[1][threadIdx.x] = sxx;
    red[2][threadIdx.x] = sc;
    __syncthreads();
    for (int s = 128; s > 0; s >>= 1) {
        if (threadIdx.x < s) {
            red[0][threadIdx.x] += red[0][threadIdx.x + s];
            red[1][threadIdx.x] += red[1][threadIdx.x + s];
            red[2][threadIdx.x] += red[2][threadIdx.x + s];
        }
        __syncthreads();
    }
    if (threadIdx.x == 0) {
        sumx[bc]      = red[0][0];
        sumx2[bc]     = red[1][0];
        sumc[bc]      = red[2][0];
        gmix_zero[bc] = 0.f;
    }
}

// ---------------------------------------------------------------------------
// K2: gate network, f16 WMMA.  One workgroup = (b, 128-wide L tile), 8 waves.
//   GEMM1: h[128,128] = relu(gw1[128,512] @ c_tile[512,128] + gb1)  -> LDS f16
//   GEMM2: g[512,128] = sigmoid(gw2[512,128] @ h + gb2); column-sum -> gmix
//
// Fragment layouts per CDNA5 ISA 7.12.2 (wave32), hf = lane/16, l16 = lane%16:
//   A 16x32 f16 : M = l16; element e -> K = (e<8 ? e : e+8) + hf*8
//                 => two contiguous 8-half runs at K = hf*8 and K = 16+hf*8
//   B 32x16 f16 : N = l16; element e -> K = (e<8 ? e : e+8) + hf*8 (same as A)
//                 => two contiguous 8-half runs at K = hf*8 and K = 16+hf*8
//   C/D 16x16 f32: VGPR r -> M = r + hf*8, N = l16
//
// LDS tiles stored so each fragment is fetched with aligned ds_load_b128:
//   aT[row][k] (K-major rows), bT[col][k], hT[col][k] (transposed).
// Rows padded to 40 / 136 halfs: 16B-aligned starts, conflict-free b128.
// Staging writes are packed (b32 pairs); hT writeback is one b128 per n-tile.
// ---------------------------------------------------------------------------
__launch_bounds__(256)
__global__ void duan_gate_gemm_kernel(const float* __restrict__ cmat,
                                      const float* __restrict__ gw1,
                                      const float* __restrict__ gb1,
                                      const float* __restrict__ gw2,
                                      const float* __restrict__ gb2,
                                      float* __restrict__ gmix_sum) {
    alignas(16) __shared__ _Float16 aT[TH][40];       // gw1 slice   (10 KB)
    alignas(16) __shared__ _Float16 bT[LT][40];       // c slice^T   (10 KB)
    alignas(16) __shared__ _Float16 hT[LT][TH + 8];   // h^T         (34 KB)

    const int b     = blockIdx.y;
    const int lbase = blockIdx.x * LT;
    const int tid   = threadIdx.x;
    const int wave  = tid >> 5;
    const int lane  = tid & 31;
    const int hf    = lane >> 4;
    const int l16   = lane & 15;

    const float* cb = cmat + (size_t)b * TC * TL;

    // ---- GEMM1: wave w owns H-rows [16w,16w+16) x 128 L columns ----
    v8f acc[8];
#pragma unroll
    for (int nt = 0; nt < 8; ++nt)
#pragma unroll
        for (int r = 0; r < 8; ++r) acc[nt][r] = 0.f;

    for (int k0 = 0; k0 < TC; k0 += 32) {
        // stage gw1[0:128, k0:k0+32] -> aT (row-major f16, packed pair writes)
        for (int i = tid; i < TH * 16; i += 256) {
            int r = i >> 4, kk = (i & 15) * 2;
            h2 p;
            p[0] = (_Float16)gw1[r * TC + k0 + kk];
            p[1] = (_Float16)gw1[r * TC + k0 + kk + 1];
            *(h2*)&aT[r][kk] = p;
        }
        // stage c[b, k0:k0+32, lbase:lbase+128] -> bT transposed
        // (two coalesced row streams, packed k-pair b32 writes)
        for (int i = tid; i < 16 * LT; i += 256) {
            int kp = i >> 7, cc = i & (LT - 1);
            const float* src = cb + (size_t)(k0 + 2 * kp) * TL + lbase + cc;
            h2 p;
            p[0] = (_Float16)src[0];
            p[1] = (_Float16)src[TL];
            *(h2*)&bT[cc][2 * kp] = p;
        }
        __syncthreads();

        // A fragment: two aligned b128 LDS loads
        v8h alo = *(const v8h*)&aT[wave * 16 + l16][hf * 8];
        v8h ahi = *(const v8h*)&aT[wave * 16 + l16][16 + hf * 8];
        v16h a = __builtin_shufflevector(alo, ahi,
                    0, 1, 2, 3, 4, 5, 6, 7, 8, 9, 10, 11, 12, 13, 14, 15);
#pragma unroll
        for (int nt = 0; nt < 8; ++nt) {
            const v8h* bp = (const v8h*)&bT[nt * 16 + l16][hf * 8];
            v16h bb = __builtin_shufflevector(bp[0], bp[2],
                        0, 1, 2, 3, 4, 5, 6, 7, 8, 9, 10, 11, 12, 13, 14, 15);
            acc[nt] = __builtin_amdgcn_wmma_f32_16x16x32_f16(
                false, a, false, bb, (short)0, acc[nt], false, false);
        }
        __syncthreads();
    }

    // bias + relu -> hT: lane's 8 rows are contiguous in hT's k-dim,
    // so each n-tile writes one aligned b128.
    {
        const float4 g0 = *(const float4*)&gb1[wave * 16 + hf * 8];
        const float4 g1 = *(const float4*)&gb1[wave * 16 + hf * 8 + 4];
        float gbv[8] = {g0.x, g0.y, g0.z, g0.w, g1.x, g1.y, g1.z, g1.w};
#pragma unroll
        for (int nt = 0; nt < 8; ++nt) {
            v8h hv;
#pragma unroll
            for (int r = 0; r < 8; ++r)
                hv[r] = (_Float16)fmaxf(acc[nt][r] + gbv[r], 0.f);
            *(v8h*)&hT[nt * 16 + l16][wave * 16 + hf * 8] = hv;
        }
    }
    __syncthreads();

    // ---- GEMM2: wave w owns C-rows [64w,64w+64), 4 m-tiles ----
    for (int mt = 0; mt < 4; ++mt) {
        const int rowbase = wave * 64 + mt * 16;
        const float* grow = gw2 + (size_t)(rowbase + l16) * TH;
        v8f acc2[8];
#pragma unroll
        for (int nt = 0; nt < 8; ++nt)
#pragma unroll
            for (int r = 0; r < 8; ++r) acc2[nt][r] = 0.f;

        for (int k0 = 0; k0 < TH; k0 += 32) {
            // A fragment from global gw2 (L2-resident): 4x float4 + cvt
            float4 w0 = *(const float4*)(grow + k0 + hf * 8);
            float4 w1 = *(const float4*)(grow + k0 + hf * 8 + 4);
            float4 w2 = *(const float4*)(grow + k0 + 16 + hf * 8);
            float4 w3 = *(const float4*)(grow + k0 + 16 + hf * 8 + 4);
            v16h a;
            a[0]  = (_Float16)w0.x; a[1]  = (_Float16)w0.y;
            a[2]  = (_Float16)w0.z; a[3]  = (_Float16)w0.w;
            a[4]  = (_Float16)w1.x; a[5]  = (_Float16)w1.y;
            a[6]  = (_Float16)w1.z; a[7]  = (_Float16)w1.w;
            a[8]  = (_Float16)w2.x; a[9]  = (_Float16)w2.y;
            a[10] = (_Float16)w2.z; a[11] = (_Float16)w2.w;
            a[12] = (_Float16)w3.x; a[13] = (_Float16)w3.y;
            a[14] = (_Float16)w3.z; a[15] = (_Float16)w3.w;
#pragma unroll
            for (int nt = 0; nt < 8; ++nt) {
                const v8h* hp = (const v8h*)&hT[nt * 16 + l16][k0 + hf * 8];
                v16h bb = __builtin_shufflevector(hp[0], hp[2],
                            0, 1, 2, 3, 4, 5, 6, 7, 8, 9, 10, 11, 12, 13, 14, 15);
                acc2[nt] = __builtin_amdgcn_wmma_f32_16x16x32_f16(
                    false, a, false, bb, (short)0, acc2[nt], false, false);
            }
        }
        // sigmoid + column reduction into gmix
#pragma unroll
        for (int r = 0; r < 8; ++r) {
            int row = rowbase + r + hf * 8;
            float bias = gb2[row];
            float s = 0.f;
#pragma unroll
            for (int nt = 0; nt < 8; ++nt) {
                float v = acc2[nt][r] + bias;
                s += 1.f / (1.f + __expf(-v));
            }
            for (int m = 8; m >= 1; m >>= 1) s += __shfl_xor(s, m, 32);
            if (l16 == 0) atomicAdd(&gmix_sum[b * TC + row], s);
        }
    }
}

// ---------------------------------------------------------------------------
// K3: per-sample stats + tiny MLP (cond MLP) + fold everything into
//     per-(b,c) affine: y = scale*x + shift
// ---------------------------------------------------------------------------
__launch_bounds__(256)
__global__ void duan_affine_kernel(const float* __restrict__ sumx,
                                   const float* __restrict__ sumx2,
                                   const float* __restrict__ sumc,
                                   const float* __restrict__ gmix_sum,
                                   const float* __restrict__ mw1,
                                   const float* __restrict__ mb1,
                                   const float* __restrict__ mw2,
                                   const float* __restrict__ mb2,
                                   float* __restrict__ scale,
                                   float* __restrict__ shift) {
    __shared__ float hm[TB][TH];
    __shared__ float gb_s[TB][2 * TC];   // gamma_beta (32 KB)
    __shared__ float muL[TB], sigL[TB];
    const int tid = threadIdx.x;
    const float invL = 1.f / (float)TL;

    if (tid < TB) {
        float sx = 0.f, sxx = 0.f;
        for (int ch = 0; ch < TC; ++ch) {
            sx  += sumx[tid * TC + ch];
            sxx += sumx2[tid * TC + ch];
        }
        float n   = (float)TC * (float)TL;
        float mu  = sx / n;
        float var = sxx / n - mu * mu;
        muL[tid]  = mu;
        sigL[tid] = sqrtf(var + 1e-5f);
    }

    // hm = relu(mw1 @ cond_pool + mb1)
    for (int i = tid; i < TB * TH; i += 256) {
        int b = i / TH, h = i % TH;
        float acc = mb1[h];
        const float* w = mw1 + h * TC;
        const float* p = sumc + b * TC;
        for (int ch = 0; ch < TC; ++ch) acc += w[ch] * (p[ch] * invL);
        hm[b][h] = fmaxf(acc, 0.f);
    }
    __syncthreads();

    // gamma_beta = mw2 @ hm + mb2
    for (int i = tid; i < TB * 2 * TC; i += 256) {
        int b = i / (2 * TC), o = i % (2 * TC);
        float acc = mb2[o];
        const float* w = mw2 + o * TH;
        for (int h = 0; h < TH; ++h) acc += w[h] * hm[b][h];
        gb_s[b][o] = acc;
    }
    __syncthreads();

    // fold into per-channel affine
    for (int i = tid; i < TB * TC; i += 256) {
        int b = i / TC, ch = i % TC;
        float muc  = sumx[i] * invL;
        float varc = sumx2[i] * invL - muc * muc;
        float sigc = sqrtf(varc + 1e-5f);
        float g    = gmix_sum[i] * invL;
        float mu   = g * muc + (1.f - g) * muL[b];
        float sg   = g * sigc + (1.f - g) * sigL[b];
        float gamma = gb_s[b][ch];
        float beta  = gb_s[b][TC + ch];
        float sc = (1.f + gamma) / sg;
        scale[i] = sc;
        shift[i] = beta - mu * sc;
    }
}

// ---------------------------------------------------------------------------
// K4: per-(b,c) sum of |scale*x + shift| over L (ranking metric)
// ---------------------------------------------------------------------------
__launch_bounds__(256)
__global__ void duan_imp_kernel(const float* __restrict__ x,
                                const float* __restrict__ scale,
                                const float* __restrict__ shift,
                                float* __restrict__ imp) {
    const int bc = blockIdx.y * TC + blockIdx.x;
    const float sc = scale[bc], sh = shift[bc];
    const float4* x4 = (const float4*)(x + (size_t)bc * TL);
    float s = 0.f;
    for (int i = threadIdx.x; i < TL / 4; i += 256) {
        __builtin_prefetch(&x4[i + 512], 0, 1);
        float4 v = x4[i];
        s += fabsf(sc * v.x + sh) + fabsf(sc * v.y + sh) +
             fabsf(sc * v.z + sh) + fabsf(sc * v.w + sh);
    }
    __shared__ float red[256];
    red[threadIdx.x] = s;
    __syncthreads();
    for (int st = 128; st > 0; st >>= 1) {
        if (threadIdx.x < st) red[threadIdx.x] += red[threadIdx.x + st];
        __syncthreads();
    }
    if (threadIdx.x == 0) imp[bc] = red[0];
}

// ---------------------------------------------------------------------------
// K5: top-KKEEP channel selection per sample (rank by counting); zero the
//     affine of dropped channels so y comes out exactly 0 there.
// ---------------------------------------------------------------------------
__launch_bounds__(256)
__global__ void duan_mask_kernel(const float* __restrict__ imp,
                                 float* __restrict__ scale,
                                 float* __restrict__ shift) {
    __shared__ float v[TC];
    const int b = blockIdx.x;
    for (int i = threadIdx.x; i < TC; i += 256) v[i] = imp[b * TC + i];
    __syncthreads();
    for (int i = threadIdx.x; i < TC; i += 256) {
        float mine = v[i];
        int rank = 0;
        for (int j = 0; j < TC; ++j) rank += (v[j] > mine) ? 1 : 0;
        if (rank >= KKEEP) {
            scale[b * TC + i] = 0.f;
            shift[b * TC + i] = 0.f;
        }
    }
}

// ---------------------------------------------------------------------------
// K6: streaming output  y = scale*x + shift (masked channels -> 0)
// ---------------------------------------------------------------------------
__launch_bounds__(256)
__global__ void duan_output_kernel(const float* __restrict__ x,
                                   const float* __restrict__ scale,
                                   const float* __restrict__ shift,
                                   float* __restrict__ out) {
    const size_t n4 = (size_t)TB * TC * TL / 4;
    const size_t stride = (size_t)gridDim.x * blockDim.x;
    for (size_t i = (size_t)blockIdx.x * blockDim.x + threadIdx.x; i < n4; i += stride) {
        __builtin_prefetch(&((const float4*)x)[i + stride], 0, 1);
        size_t bc = (i * 4) / TL;
        float sc = scale[bc], sh = shift[bc];
        float4 v = ((const float4*)x)[i];
        float4 r;
        r.x = sc * v.x + sh;
        r.y = sc * v.y + sh;
        r.z = sc * v.z + sh;
        r.w = sc * v.w + sh;
        ((float4*)out)[i] = r;
    }
}

// ---------------------------------------------------------------------------
extern "C" void kernel_launch(void* const* d_in, const int* in_sizes, int n_in,
                              void* d_out, int out_size, void* d_ws, size_t ws_size,
                              hipStream_t stream) {
    const float* x    = (const float*)d_in[0];
    const float* cmat = (const float*)d_in[1];
    const float* gw1  = (const float*)d_in[2];
    const float* gb1  = (const float*)d_in[3];
    const float* gw2  = (const float*)d_in[4];
    const float* gb2  = (const float*)d_in[5];
    const float* mw1  = (const float*)d_in[6];
    const float* mb1  = (const float*)d_in[7];
    const float* mw2  = (const float*)d_in[8];
    const float* mb2  = (const float*)d_in[9];
    float* out = (float*)d_out;

    float* ws    = (float*)d_ws;            // 28672 floats = 112 KB used
    float* sumx  = ws;
    float* sumx2 = ws + 4096;
    float* sumc  = ws + 8192;
    float* gmix  = ws + 12288;
    float* scale = ws + 16384;
    float* shift = ws + 20480;
    float* imp   = ws + 24576;

    dim3 gridBC(TC, TB);                    // (512, 8)
    duan_stats_kernel<<<gridBC, 256, 0, stream>>>(x, cmat, sumx, sumx2, sumc, gmix);
    duan_gate_gemm_kernel<<<dim3(TL / LT, TB), 256, 0, stream>>>(cmat, gw1, gb1,
                                                                 gw2, gb2, gmix);
    duan_affine_kernel<<<1, 256, 0, stream>>>(sumx, sumx2, sumc, gmix,
                                              mw1, mb1, mw2, mb2, scale, shift);
    duan_imp_kernel<<<gridBC, 256, 0, stream>>>(x, scale, shift, imp);
    duan_mask_kernel<<<TB, 256, 0, stream>>>(imp, scale, shift);
    duan_output_kernel<<<8192, 256, 0, stream>>>(x, scale, shift, out);
}
